// GATGAPModel_23098334117974
// MI455X (gfx1250) — matrix operand order repaired and checked by reading, not compiled, not run on hardware
//
#include <hip/hip_runtime.h>
#include <hip/hip_bf16.h>

// ---------------- problem constants ----------------
#define NN     50000
#define EE     400000
#define GG     1000
#define HEADS  4
#define UNITS  128
#define HU     512          // HEADS*UNITS
#define NFEAT  8
#define EMBW   8
#define MT     (NN/16)      // 3125 row tiles (N divisible by 16)
#define KT_FULL (HU/32)     // 16 k-steps for K=512

typedef __attribute__((ext_vector_type(16))) __bf16 v16bf;
typedef __attribute__((ext_vector_type(8)))  float  v8f;

union Frag { v16bf v; uint4 u[2]; };

__device__ __forceinline__ float leakyf(float x, float s) { return x > 0.f ? x : s * x; }
__device__ __forceinline__ void load_frag(const uint4* __restrict__ p, Frag& f) {
    f.u[0] = p[0]; f.u[1] = p[1];
}

// ---------------- small utility kernels ----------------
__global__ void fill_neg_inf(float* __restrict__ p, int n) {
    int t = blockIdx.x * blockDim.x + threadIdx.x;
    if (t < n) p[t] = __uint_as_float(0xFF800000u);
}

__global__ void count_nodes(const int* __restrict__ gid, int* __restrict__ counts) {
    int t = blockIdx.x * blockDim.x + threadIdx.x;
    if (t < NN) atomicAdd(&counts[gid[t]], 1);
}

// h0[n, 0:8] = emb[x_pdg[n]], h0[n, 8:16] = x_feature[n]
__global__ void build_h0(const float* __restrict__ emb, const int* __restrict__ x_pdg,
                         const float* __restrict__ x_feat, float* __restrict__ h0) {
    int t = blockIdx.x * blockDim.x + threadIdx.x;
    if (t >= NN * 16) return;
    int n = t >> 4, k = t & 15;
    h0[t] = (k < EMBW) ? emb[x_pdg[n] * EMBW + k] : x_feat[n * NFEAT + (k - EMBW)];
}

// ---------------- WMMA fragment packing ----------------
// A fragment (16x32 bf16): lane L holds row m = (L&15), kg = L>>4;
//   element i (0..15): k = ktile*32 + ((i<8)? i : i+8) + kg*8
__global__ void pack_A(const float* __restrict__ src, __bf16* __restrict__ dst,
                       int Ksrc, int KT, int mtiles) {
    int t = blockIdx.x * blockDim.x + threadIdx.x;
    if (t >= mtiles * KT * 32) return;
    int lane = t & 31;
    int kt   = (t >> 5) % KT;
    int mt   = (t >> 5) / KT;
    int m    = mt * 16 + (lane & 15);
    int kg   = lane >> 4;
    __bf16 vals[16];
#pragma unroll
    for (int i = 0; i < 16; ++i) {
        int k = kt * 32 + ((i < 8) ? i : i + 8) + kg * 8;
        float v = (k < Ksrc) ? src[(size_t)m * Ksrc + k] : 0.f;
        vals[i] = (__bf16)v;
    }
    uint4 u0, u1;
    __builtin_memcpy(&u0, &vals[0], 16);
    __builtin_memcpy(&u1, &vals[8], 16);
    uint4* o = (uint4*)dst + (size_t)t * 2;
    o[0] = u0; o[1] = u1;
}

// B fragment (32x16 bf16): lane L holds col n = ntile*16 + (L&15), kg = L>>4;
//   element i (0..15): k = ktile*32 + ((i<8)? i : i+8) + kg*8.  W is [K, 512] row-major.
__global__ void pack_B(const float* __restrict__ W, __bf16* __restrict__ dst,
                       int Ksrc, int KT) {
    int t = blockIdx.x * blockDim.x + threadIdx.x;
    if (t >= KT * 32 * 32) return;
    int lane = t & 31;
    int nt   = (t >> 5) & 31;
    int kt   = t >> 10;
    int n    = nt * 16 + (lane & 15);
    int kg   = lane >> 4;
    __bf16 vals[16];
#pragma unroll
    for (int i = 0; i < 16; ++i) {
        int k = kt * 32 + ((i < 8) ? i : i + 8) + kg * 8;
        float v = (k < Ksrc) ? W[(size_t)k * HU + n] : 0.f;
        vals[i] = (__bf16)v;
    }
    uint4 u0, u1;
    __builtin_memcpy(&u0, &vals[0], 16);
    __builtin_memcpy(&u1, &vals[8], 16);
    uint4* o = (uint4*)dst + (size_t)t * 2;
    o[0] = u0; o[1] = u1;
}

// ---------------- GEMM: z = Apk @ Bpk, f32 out [NN, 512] ----------------
// one wave -> 16x64 tile (4 n-tiles), 8 waves per block -> one block per mtile.
// KT is a compile-time constant so the k-loop fully unrolls: straight-line
// code lets the scheduler hoist global_load_b128 across earlier WMMAs
// (register-level pipelining) with no accumulator copies.
template <int KT>
__global__ __launch_bounds__(256)
void gemm_wmma_bf16(const __bf16* __restrict__ Apk, const __bf16* __restrict__ Bpk,
                    float* __restrict__ Cout, int mtiles) {
    const int lane   = threadIdx.x & 31;
    const int wave   = threadIdx.x >> 5;
    const int gw     = blockIdx.x * 8 + wave;
    const int mt     = gw >> 3;
    const int nchunk = gw & 7;          // 8 chunks of 64 columns
    if (mt >= mtiles) return;

    v8f acc[4];
#pragma unroll
    for (int j = 0; j < 4; ++j) acc[j] = (v8f){0.f,0.f,0.f,0.f,0.f,0.f,0.f,0.f};

    const uint4* aBase = (const uint4*)Apk + (size_t)mt * KT * 64 + lane * 2;
    const uint4* bBase = (const uint4*)Bpk + (size_t)(nchunk * 4) * 64 + lane * 2;
    // per k-step: A frag at aBase + kt*64 ; B frag j at bBase + (kt*32 + j)*64

#pragma unroll
    for (int kt = 0; kt < KT; ++kt) {
        Frag a;
        load_frag(aBase + (size_t)kt * 64, a);
        Frag b[4];
#pragma unroll
        for (int j = 0; j < 4; ++j)
            load_frag(bBase + ((size_t)kt * 32 + j) * 64, b[j]);
#pragma unroll
        for (int j = 0; j < 4; ++j)
            acc[j] = __builtin_amdgcn_wmma_f32_16x16x32_bf16(
                false, a.v, false, b[j].v, (short)0, acc[j], false, false);
    }

    const int kg = lane >> 4, nloc = lane & 15;
#pragma unroll
    for (int j = 0; j < 4; ++j)
#pragma unroll
        for (int r = 0; r < 8; ++r) {
            int m = mt * 16 + r + 8 * kg;
            int n = nchunk * 64 + j * 16 + nloc;
            Cout[(size_t)m * HU + n] = acc[j][r];
        }
}

// ---------------- attention scores: el/er = <z[n,h,:], attn> ----------------
__global__ __launch_bounds__(256)
void eler_kernel(const float* __restrict__ z, const float* __restrict__ attn_l,
                 const float* __restrict__ attn_r, float* __restrict__ el,
                 float* __restrict__ er) {
    int lane = threadIdx.x & 31;
    int w    = blockIdx.x * 8 + (threadIdx.x >> 5);   // (n,h) index
    if (w >= NN * HEADS) return;
    int h = w & 3, n = w >> 2;
    float4 zv = *((const float4*)(z + (size_t)n * HU + h * UNITS) + lane);
    float4 lv = *((const float4*)(attn_l + h * UNITS) + lane);
    float4 rv = *((const float4*)(attn_r + h * UNITS) + lane);
    float sl = zv.x*lv.x + zv.y*lv.y + zv.z*lv.z + zv.w*lv.w;
    float sr = zv.x*rv.x + zv.y*rv.y + zv.z*rv.z + zv.w*rv.w;
#pragma unroll
    for (int off = 16; off > 0; off >>= 1) {
        sl += __shfl_down(sl, off, 32);
        sr += __shfl_down(sr, off, 32);
    }
    if (lane == 0) { el[w] = sl; er[w] = sr; }
}

// ---------------- edge softmax passes ----------------
__global__ void edge_max(const int* __restrict__ src, const int* __restrict__ dst,
                         const float* __restrict__ el, const float* __restrict__ er,
                         float* __restrict__ emax) {
    int t = blockIdx.x * blockDim.x + threadIdx.x;
    if (t >= EE * HEADS) return;
    int h = t & 3, e = t >> 2;
    int s = src[e], d = dst[e];
    float v = leakyf(el[s * 4 + h] + er[d * 4 + h], 0.2f);
    float* addr = &emax[d * 4 + h];
    if (v >= 0.f) atomicMax((int*)addr, __float_as_int(v));
    else          atomicMin((unsigned int*)addr, __float_as_uint(v));
}

__global__ void edge_exp(const int* __restrict__ src, const int* __restrict__ dst,
                         const float* __restrict__ el, const float* __restrict__ er,
                         const float* __restrict__ emax, float* __restrict__ abuf,
                         float* __restrict__ denom) {
    int t = blockIdx.x * blockDim.x + threadIdx.x;
    if (t >= EE * HEADS) return;
    int h = t & 3, e = t >> 2;
    int s = src[e], d = dst[e];
    float v = leakyf(el[s * 4 + h] + er[d * 4 + h], 0.2f);
    float a = __expf(v - emax[d * 4 + h]);
    abuf[t] = a;
    atomicAdd(&denom[d * 4 + h], a);
}

// out[dst, h, :] += alpha * z[src, h, :]  (4 units per thread)
__global__ __launch_bounds__(256)
void edge_aggr(const int* __restrict__ src, const int* __restrict__ dst,
               const float* __restrict__ abuf, const float* __restrict__ denom,
               const float* __restrict__ z, float* __restrict__ out) {
    int t = blockIdx.x * blockDim.x + threadIdx.x;   // < EE * 128
    if (t >= EE * 128) return;
    int e = t >> 7, q = t & 127;
    int h = q >> 5, u0 = (q & 31) * 4;
    int s = src[e], d = dst[e];
    float alpha = abuf[(size_t)e * 4 + h] / fmaxf(denom[d * 4 + h], 1e-9f);
    float4 zv = *(const float4*)(z + (size_t)s * HU + h * UNITS + u0);
    float* o = out + (size_t)d * HU + h * UNITS + u0;
    atomicAdd(o + 0, alpha * zv.x);
    atomicAdd(o + 1, alpha * zv.y);
    atomicAdd(o + 2, alpha * zv.z);
    atomicAdd(o + 3, alpha * zv.w);
}

// h = leaky(out + gat_b, 0.01), also accumulate graph sums
__global__ void finalize_h(float* __restrict__ out, const float* __restrict__ gat_b,
                           const int* __restrict__ gid, float* __restrict__ hsum) {
    int t = blockIdx.x * blockDim.x + threadIdx.x;
    if (t >= NN * HU) return;
    int n = t >> 9, k = t & 511;
    float v = leakyf(out[t] + gat_b[k], 0.01f);
    out[t] = v;
    atomicAdd(&hsum[(size_t)gid[n] * HU + k], v);
}

// hg_new = leaky([hg_prev || hmean] @ fc_W + fc_b, 0.01); block per graph, thread per out col
__global__ __launch_bounds__(128)
void fc_kernel(const float* __restrict__ hgprev, const float* __restrict__ hsum,
               const int* __restrict__ counts, const float* __restrict__ fcW,
               const float* __restrict__ fcb, float* __restrict__ hgnew, int in_g) {
    int g = blockIdx.x, j = threadIdx.x;
    float invc = 1.f / fmaxf((float)counts[g], 1.f);
    float acc = fcb[j];
    for (int i = 0; i < in_g; ++i)
        acc += hgprev[g * UNITS + i] * fcW[(size_t)i * UNITS + j];
    for (int i = 0; i < HU; ++i)
        acc += hsum[(size_t)g * HU + i] * invc * fcW[(size_t)(in_g + i) * UNITS + j];
    hgnew[g * UNITS + j] = leakyf(acc, 0.01f);
}

__global__ void out_kernel(const float* __restrict__ hg, const float* __restrict__ outW,
                           const float* __restrict__ outb, float* __restrict__ y) {
    int g = blockIdx.x * blockDim.x + threadIdx.x;
    if (g >= GG) return;
    float acc = outb[0];
    for (int i = 0; i < UNITS; ++i) acc += hg[g * UNITS + i] * outW[i];
    y[g] = acc;
}

// ---------------- host driver ----------------
extern "C" void kernel_launch(void* const* d_in, const int* in_sizes, int n_in,
                              void* d_out, int out_size, void* d_ws, size_t ws_size,
                              hipStream_t stream) {
    (void)in_sizes; (void)n_in; (void)out_size; (void)ws_size;
    // Flattened input order (insertion order of reference dicts):
    // 0: emb; per layer i (base 1+6i): W, attn_l, attn_r, gat_b, fc_W, fc_b;
    // 31: out_W; 32: out_b; 33: x_feature; 34: x_pdg; 35: src; 36: dst; 37: graph_id
    const float* emb = (const float*)d_in[0];
    struct Layer { const float *W, *attn_l, *attn_r, *gat_b, *fc_W, *fc_b; };
    Layer L[5];
    for (int i = 0; i < 5; ++i) {
        int b = 1 + 6 * i;
        L[i].W      = (const float*)d_in[b + 0];
        L[i].attn_l = (const float*)d_in[b + 1];
        L[i].attn_r = (const float*)d_in[b + 2];
        L[i].gat_b  = (const float*)d_in[b + 3];
        L[i].fc_W   = (const float*)d_in[b + 4];
        L[i].fc_b   = (const float*)d_in[b + 5];
    }
    const float* outW  = (const float*)d_in[31];
    const float* outb  = (const float*)d_in[32];
    const float* xfeat = (const float*)d_in[33];
    const int*   xpdg  = (const int*)d_in[34];
    const int*   src   = (const int*)d_in[35];
    const int*   dst   = (const int*)d_in[36];
    const int*   gid   = (const int*)d_in[37];

    // workspace carve
    size_t off = 0;
    auto carve = [&](size_t bytes) -> void* {
        void* p = (char*)d_ws + off;
        off += (bytes + 255) & ~(size_t)255;
        return p;
    };
    float*  zbuf   = (float*)carve((size_t)NN * HU * 4);          // 102.4 MB
    float*  outbuf = (float*)carve((size_t)NN * HU * 4);          // 102.4 MB
    __bf16* Apk    = (__bf16*)carve((size_t)MT * KT_FULL * 1024); // 51.2 MB
    __bf16* Bpk    = (__bf16*)carve((size_t)KT_FULL * 32 * 1024); // 512 KB
    float*  h0     = (float*)carve((size_t)NN * 16 * 4);
    float*  el     = (float*)carve((size_t)NN * HEADS * 4);
    float*  er     = (float*)carve((size_t)NN * HEADS * 4);
    float*  emax   = (float*)carve((size_t)NN * HEADS * 4);
    float*  denom  = (float*)carve((size_t)NN * HEADS * 4);
    float*  abuf   = (float*)carve((size_t)EE * HEADS * 4);
    float*  hsum   = (float*)carve((size_t)GG * HU * 4);
    float*  hgA    = (float*)carve((size_t)GG * UNITS * 4);
    float*  hgB    = (float*)carve((size_t)GG * UNITS * 4);
    int*    counts = (int*)carve((size_t)GG * 4);

    // graph node counts
    hipMemsetAsync(counts, 0, (size_t)GG * 4, stream);
    count_nodes<<<(NN + 255) / 256, 256, 0, stream>>>(gid, counts);

    // layer-0 input features -> packed A (K=16 padded to 32)
    build_h0<<<(NN * 16) / 256, 256, 0, stream>>>(emb, xpdg, xfeat, h0);
    pack_A<<<(MT * 1 * 32 + 255) / 256, 256, 0, stream>>>(h0, Apk, 16, 1, MT);

    float* hg_prev = hgA;
    float* hg_next = hgB;
    for (int l = 0; l < 5; ++l) {
        const int in_f = (l == 0) ? 16 : HU;
        const int KT   = (l == 0) ? 1 : KT_FULL;
        const int in_g = (l == 0) ? 0 : UNITS;

        // weights -> packed B, then GEMM: z = h @ W
        pack_B<<<(KT * 1024 + 255) / 256, 256, 0, stream>>>(L[l].W, Bpk, in_f, KT);
        if (l == 0)
            gemm_wmma_bf16<1><<<MT, 256, 0, stream>>>(Apk, Bpk, zbuf, MT);
        else
            gemm_wmma_bf16<KT_FULL><<<MT, 256, 0, stream>>>(Apk, Bpk, zbuf, MT);

        // attention logits per (node, head)
        eler_kernel<<<(NN * HEADS) / 8, 256, 0, stream>>>(zbuf, L[l].attn_l, L[l].attn_r, el, er);

        // edge softmax
        fill_neg_inf<<<(NN * HEADS + 255) / 256, 256, 0, stream>>>(emax, NN * HEADS);
        hipMemsetAsync(denom, 0, (size_t)NN * HEADS * 4, stream);
        edge_max<<<(EE * HEADS) / 256, 256, 0, stream>>>(src, dst, el, er, emax);
        edge_exp<<<(EE * HEADS) / 256, 256, 0, stream>>>(src, dst, el, er, emax, abuf, denom);

        // message aggregation
        hipMemsetAsync(outbuf, 0, (size_t)NN * HU * 4, stream);
        edge_aggr<<<(EE * 128) / 256, 256, 0, stream>>>(src, dst, abuf, denom, zbuf, outbuf);

        // h = leaky(out + gat_b); graph pooling sums
        hipMemsetAsync(hsum, 0, (size_t)GG * HU * 4, stream);
        finalize_h<<<(NN * HU) / 256, 256, 0, stream>>>(outbuf, L[l].gat_b, gid, hsum);

        // graph-level FC
        fc_kernel<<<GG, 128, 0, stream>>>(hg_prev, hsum, counts, L[l].fc_W, L[l].fc_b,
                                          hg_next, in_g);

        // pack h for the next layer's GEMM
        if (l < 4)
            pack_A<<<(MT * KT_FULL * 32 + 255) / 256, 256, 0, stream>>>(outbuf, Apk, HU, KT_FULL, MT);

        float* tmp = hg_prev; hg_prev = hg_next; hg_next = tmp;
    }

    out_kernel<<<(GG + 255) / 256, 256, 0, stream>>>(hg_prev, outW, outb, (float*)d_out);
}
